// BDHAttentionLayer_12352325943546
// MI455X (gfx1250) — matrix operand order repaired and checked by reading, not compiled, not run on hardware
//
#include <hip/hip_runtime.h>

typedef __attribute__((ext_vector_type(16))) _Float16 v16h;
typedef __attribute__((ext_vector_type(8)))  float    v8f;
typedef __attribute__((ext_vector_type(4)))  unsigned int u32x4;
typedef __attribute__((ext_vector_type(8)))  int      i32x8;
typedef __attribute__((ext_vector_type(4)))  int      i32x4;

#define B_  2
#define S_  2048
#define DM_ 1024
#define H_  16
#define DH_ 64

__device__ inline v8f zero8() {
  v8f z;
  for (int i = 0; i < 8; ++i) z[i] = 0.f;
  return z;
}

__device__ inline v8f wmma_f16(v16h a, v16h b, v8f c) {
  // (neg_a, A, neg_b, B, c_mod, C, reuse_a, reuse_b)
  return __builtin_amdgcn_wmma_f32_16x16x32_f16(false, a, false, b, (short)0, c,
                                                false, false);
}

// Build a 16-half fragment from two 16-byte chunks (global or LDS).
__device__ inline v16h load2(const _Float16* p0, const _Float16* p1) {
  union U { v16h v; uint4 q[2]; } u;
  u.q[0] = *(const uint4*)p0;
  u.q[1] = *(const uint4*)p1;
  return u.v;
}

#if __has_builtin(__builtin_amdgcn_tensor_load_to_lds)
#define USE_TDM 1
#else
#define USE_TDM 0
#endif

#if USE_TDM
// Issue a TDM 2D tile load: tile_x elements (2B each) x tile_y rows from a
// row-major tensor with row stride stride_x (elements) into LDS at lds_off.
// LDS rows are padded: 16 DWORDs data + 4 DWORDs pad -> 40-half row stride.
__device__ inline void tdm_load_2d(const void* gptr, unsigned lds_byte_off,
                                   unsigned tile_x, unsigned tile_y,
                                   unsigned stride_x) {
  unsigned long long ga = (unsigned long long)(uintptr_t)gptr;
  u32x4 g0;
  g0[0] = 1u;                                   // count=1, user descriptor
  g0[1] = lds_byte_off;                         // lds_addr (bytes)
  g0[2] = (unsigned)(ga & 0xFFFFFFFFu);         // global_addr[31:0]
  g0[3] = (unsigned)((ga >> 32) & 0x1FFFFFFu)   // global_addr[56:32]
          | (2u << 30);                         // type = 2 ("image")
  i32x8 g1;
  g1[0] = (int)((1u << 16)        // data_size = 1 -> 2 bytes
                | (1u << 20)      // pad_enable
                | (3u << 22)      // pad_interval: 16 DWORDs
                | (3u << 25));    // pad_amount: 4 DWORDs
  g1[1] = (int)((tile_x & 0xFFFFu) << 16);          // tensor_dim0[15:0]
  g1[2] = (int)((tile_x >> 16) | ((tile_y & 0xFFFFu) << 16));  // dim0 hi|dim1 lo
  g1[3] = (int)((tile_y >> 16) | (tile_x << 16));   // dim1 hi | tile_dim0
  g1[4] = (int)(tile_y & 0xFFFFu);                  // tile_dim1 (tile_dim2=0)
  g1[5] = (int)stride_x;                            // tensor_dim0_stride[31:0]
  g1[6] = 0;                                        // stride0 hi | stride1 lo
  g1[7] = 0;
  i32x4 z4 = {0, 0, 0, 0};
  i32x8 z8 = {0, 0, 0, 0, 0, 0, 0, 0};
  __builtin_amdgcn_tensor_load_to_lds(g0, g1, z4, z4, z8, 0);
}
#endif

// ---------------------------------------------------------------- conversions
__global__ void cvt_kernel(const float* __restrict__ in,
                           _Float16* __restrict__ out, int n4) {
  int i = blockIdx.x * blockDim.x + threadIdx.x;
  const int stride = gridDim.x * blockDim.x;
  const float4* in4 = (const float4*)in;
  uint2* out4 = (uint2*)out;
  for (; i < n4; i += stride) {
    const float4 f = in4[i];
    union { _Float16 h[4]; uint2 u; } o;
    o.h[0] = (_Float16)f.x;
    o.h[1] = (_Float16)f.y;
    o.h[2] = (_Float16)f.z;
    o.h[3] = (_Float16)f.w;
    out4[i] = o.u;
  }
}

// ------------------------------------------------------------------- WMMA GEMM
// C(M=4096,N=1024) = A(M,K=1024) * W(N,K)^T + bias
// MODE 0: f16 [M][N]   (Q, K projections)
// MODE 1: f16 transposed per head: [B,H,DH,S]   (V projection)
// MODE 2: f32 [M][N]   (output projection)
template <int MODE>
__global__ __launch_bounds__(256) void gemm_wmma(
    const _Float16* __restrict__ A, const _Float16* __restrict__ W,
    const float* __restrict__ bias, void* __restrict__ out) {
  const int N = DM_, K = DM_;
  // [buf][0]=A tile, [buf][1]=W tile; 128 rows x 32 halves, row stride 40.
  __shared__ _Float16 Tl[2][2][128 * 40];

  const int tid = threadIdx.x;
  const int lane = tid & 31;
  const int w = tid >> 5;           // 8 waves
  const int waveM = w & 3;          // 4 x 2 wave grid
  const int waveN = w >> 2;
  const int mBase = blockIdx.y * 128;
  const int nBase = blockIdx.x * 128;
  const int hsel = (lane >> 4) & 1;
  const int l15 = lane & 15;

  v8f acc[2][4];
  for (int i = 0; i < 2; ++i)
    for (int j = 0; j < 4; ++j) acc[i][j] = zero8();

#if USE_TDM
  const unsigned ldsA = (unsigned)(uintptr_t)(void*)&Tl[0][0][0];
  const unsigned ldsW = (unsigned)(uintptr_t)(void*)&Tl[0][1][0];
  const unsigned bufStride = 2u * 128u * 40u * 2u;  // bytes between buffers
  if (w == 0) {
    tdm_load_2d(A + (unsigned)mBase * K, ldsA, 32, 128, K);
    tdm_load_2d(W + (unsigned)nBase * K, ldsW, 32, 128, K);
  }
#else
  const int sr = tid >> 1;          // staged row 0..127
  const int sc = (tid & 1) * 16;    // half-row offset 0 or 16
#endif

  int buf = 0;
  for (int k0 = 0; k0 < K; k0 += 32) {
#if USE_TDM
    if (w == 0) __builtin_amdgcn_s_wait_tensorcnt(0);
    __syncthreads();
    if (w == 0 && k0 + 32 < K) {
      const unsigned nb = (unsigned)(buf ^ 1) * bufStride;
      tdm_load_2d(A + (unsigned)mBase * K + k0 + 32, ldsA + nb, 32, 128, K);
      tdm_load_2d(W + (unsigned)nBase * K + k0 + 32, ldsW + nb, 32, 128, K);
    }
#else
    {
      const uint4* ga = (const uint4*)(A + (unsigned)(mBase + sr) * K + k0 + sc);
      uint4* la = (uint4*)(&Tl[buf][0][0] + sr * 40 + sc);
      la[0] = ga[0];
      la[1] = ga[1];
      const uint4* gw = (const uint4*)(W + (unsigned)(nBase + sr) * K + k0 + sc);
      uint4* lw = (uint4*)(&Tl[buf][1][0] + sr * 40 + sc);
      lw[0] = gw[0];
      lw[1] = gw[1];
    }
    __syncthreads();
#endif

    const _Float16* As = &Tl[buf][0][0];
    const _Float16* Ws = &Tl[buf][1][0];
    v16h af[2], bf[4];
    for (int mt = 0; mt < 2; ++mt) {
      const _Float16* p = As + (waveM * 32 + mt * 16 + l15) * 40;
      af[mt] = load2(p + hsel * 8, p + 16 + hsel * 8);
    }
    for (int nt = 0; nt < 4; ++nt) {
      const _Float16* p = Ws + (waveN * 64 + nt * 16 + l15) * 40;
      bf[nt] = load2(p + hsel * 8, p + 16 + hsel * 8);
    }
    for (int mt = 0; mt < 2; ++mt)
      for (int nt = 0; nt < 4; ++nt)
        acc[mt][nt] = wmma_f16(af[mt], bf[nt], acc[mt][nt]);

#if !USE_TDM
    __syncthreads();
#endif
    buf ^= 1;
  }

  // Epilogue: bias + store. C layout: VGPR r -> M = r + 8*hsel, N = lane&15.
  for (int nt = 0; nt < 4; ++nt) {
    const unsigned n = nBase + waveN * 64 + nt * 16 + l15;
    const float bi = bias[n];
    for (int mt = 0; mt < 2; ++mt) {
      const unsigned m0 = mBase + waveM * 32 + mt * 16 + hsel * 8;
      for (int r = 0; r < 8; ++r) {
        const unsigned m = m0 + r;
        const float v = acc[mt][nt][r] + bi;
        if (MODE == 0) {
          ((_Float16*)out)[m * (unsigned)N + n] = (_Float16)v;
        } else if (MODE == 1) {
          const unsigned b = m >> 11, s = m & 2047;      // S_ = 2048
          const unsigned hh = n >> 6, d = n & 63;
          ((_Float16*)out)[(((b * H_ + hh) * DH_ + d) << 11) + s] = (_Float16)v;
        } else {
          ((float*)out)[m * (unsigned)N + n] = v;
        }
      }
    }
  }
}

// --------------------------------------------------------- fused ReLU attention
// Per wave: 16-row Q strip. Single pass: ctx += relu(QK^T/8 masked) * V,
// rowsum accumulated, normalized at the end. K/V stream from global (L2-hot:
// all of K,V in f16 is 16 MiB << 192 MB L2, reused 16x across q-blocks).
__global__ __launch_bounds__(256) void attn_kernel(
    const _Float16* __restrict__ Q, const _Float16* __restrict__ Kb,
    const _Float16* __restrict__ Vt, _Float16* __restrict__ Ctx) {
  __shared__ _Float16 Pb[8][16 * 40];  // per-wave P bounce (16 x 32, stride 40)

  const int lane = threadIdx.x & 31;
  const int w = threadIdx.x >> 5;
  const int bh = blockIdx.y;
  const int b = bh >> 4;   // H_ == 16
  const int h = bh & 15;
  const int q0 = blockIdx.x * 128 + w * 16;
  const int hsel = (lane >> 4) & 1;
  const int l15 = lane & 15;

  // Q fragments live in registers for the whole kernel (rows q0..q0+15).
  const _Float16* qrow = Q + (unsigned)(b * S_ + q0 + l15) * DM_ + h * DH_;
  const v16h aq0 = load2(qrow + hsel * 8, qrow + 16 + hsel * 8);
  const v16h aq1 = load2(qrow + 32 + hsel * 8, qrow + 48 + hsel * 8);

  const _Float16* kbase = Kb + (unsigned)(b * S_) * DM_ + h * DH_;
  const _Float16* vbase = Vt + (unsigned)((b * H_ + h) * DH_) * S_;

  float rs[8];
  for (int r = 0; r < 8; ++r) rs[r] = 0.f;
  v8f cacc[4];
  for (int i = 0; i < 4; ++i) cacc[i] = zero8();

  for (int kb = 0; kb <= q0 + 15; kb += 32) {
    // Prefetch next key chunk while this one computes.
    if (kb + 32 <= q0 + 15) {
      __builtin_prefetch(kbase + (unsigned)(kb + 32 + l15) * DM_, 0, 3);
      __builtin_prefetch(vbase + (unsigned)(l15 * S_) + kb + 32, 0, 3);
    }
    // ---- scores for keys [kb, kb+32): two 16x16 tiles
    for (int sub = 0; sub < 2; ++sub) {
      const int kt = kb + sub * 16;
      const _Float16* krow = kbase + (unsigned)(kt + l15) * DM_;
      const v16h bk0 = load2(krow + hsel * 8, krow + 16 + hsel * 8);
      const v16h bk1 = load2(krow + 32 + hsel * 8, krow + 48 + hsel * 8);
      v8f s_acc = zero8();
      s_acc = wmma_f16(aq0, bk0, s_acc);
      s_acc = wmma_f16(aq1, bk1, s_acc);

      const int key = kt + l15;
      for (int r = 0; r < 8; ++r) {
        const int m = q0 + r + hsel * 8;
        float v = s_acc[r] * 0.125f;                 // 1/sqrt(DH)
        v = (key <= m && v > 0.f) ? v : 0.f;         // causal + relu
        rs[r] += v;
        Pb[w][(r + hsel * 8) * 40 + sub * 16 + l15] = (_Float16)v;
      }
    }
    asm volatile("s_wait_dscnt 0" ::: "memory");  // per-wave LDS RAW fence

    // P re-read as 16x32 A-fragment
    const _Float16* pr = &Pb[w][l15 * 40];
    const v16h ap = load2(pr + hsel * 8, pr + 16 + hsel * 8);

    // ctx += P * V  (V transposed: contiguous over keys)
    for (int nt = 0; nt < 4; ++nt) {
      const _Float16* vp = vbase + (unsigned)((nt * 16 + l15) * S_) + kb;
      const v16h bv = load2(vp + hsel * 8, vp + 16 + hsel * 8);
      cacc[nt] = wmma_f16(ap, bv, cacc[nt]);
    }
  }

  // Row-sum reduction across the 16 lanes sharing each M row.
  for (int r = 0; r < 8; ++r) {
    float v = rs[r];
    v += __shfl_xor(v, 1, 32);
    v += __shfl_xor(v, 2, 32);
    v += __shfl_xor(v, 4, 32);
    v += __shfl_xor(v, 8, 32);
    rs[r] = 1.0f / (v + 1e-9f);
  }
  _Float16* crow = Ctx + (unsigned)(b * S_ + q0 + hsel * 8) * DM_ + h * DH_;
  for (int nt = 0; nt < 4; ++nt) {
    for (int r = 0; r < 8; ++r) {
      crow[(unsigned)(r * DM_) + nt * 16 + l15] = (_Float16)(cacc[nt][r] * rs[r]);
    }
  }
}

// -------------------------------------------------------------------- launcher
extern "C" void kernel_launch(void* const* d_in, const int* in_sizes, int n_in,
                              void* d_out, int out_size, void* d_ws,
                              size_t ws_size, hipStream_t stream) {
  (void)in_sizes; (void)n_in; (void)out_size; (void)ws_size;
  const float* x  = (const float*)d_in[0];
  const float* Wq = (const float*)d_in[1];
  const float* bq = (const float*)d_in[2];
  const float* Wk = (const float*)d_in[3];
  const float* bk = (const float*)d_in[4];
  const float* Wv = (const float*)d_in[5];
  const float* bv = (const float*)d_in[6];
  const float* Wo = (const float*)d_in[7];
  const float* bo = (const float*)d_in[8];
  float* out = (float*)d_out;

  const size_t MK = (size_t)B_ * S_ * DM_;  // 4M halves
  const size_t WW = (size_t)DM_ * DM_;      // 1M halves
  _Float16* p = (_Float16*)d_ws;
  _Float16* xh  = p; p += MK;
  _Float16* wqh = p; p += WW;
  _Float16* wkh = p; p += WW;
  _Float16* wvh = p; p += WW;
  _Float16* woh = p; p += WW;
  _Float16* qb  = p; p += MK;
  _Float16* kbf = p; p += MK;
  _Float16* vt  = p; p += MK;
  _Float16* ctx = p; p += MK;
  // total ws use: 48 MiB

  cvt_kernel<<<512, 256, 0, stream>>>(x,  xh,  (int)(MK / 4));
  cvt_kernel<<<256, 256, 0, stream>>>(Wq, wqh, (int)(WW / 4));
  cvt_kernel<<<256, 256, 0, stream>>>(Wk, wkh, (int)(WW / 4));
  cvt_kernel<<<256, 256, 0, stream>>>(Wv, wvh, (int)(WW / 4));
  cvt_kernel<<<256, 256, 0, stream>>>(Wo, woh, (int)(WW / 4));

  dim3 gg(DM_ / 128, (B_ * S_) / 128);  // (8, 32)
  gemm_wmma<0><<<gg, 256, 0, stream>>>(xh, wqh, bq, qb);
  gemm_wmma<0><<<gg, 256, 0, stream>>>(xh, wkh, bk, kbf);
  gemm_wmma<1><<<gg, 256, 0, stream>>>(xh, wvh, bv, vt);

  attn_kernel<<<dim3(S_ / 128, B_ * H_), 256, 0, stream>>>(qb, kbf, vt, ctx);

  gemm_wmma<2><<<gg, 256, 0, stream>>>(ctx, woh, bo, out);
}
